// SA_61272003445142
// MI455X (gfx1250) — compile-verified, matrix operand order checked by reading, not run on hardware
//
#include <hip/hip_runtime.h>

// ---------------------------------------------------------------------------
// Self-attention forward for MI455X (gfx1250), bf16 WMMA pipeline with
// async global->LDS staging (ASYNCcnt) for shared operands.
// B=4, S=2048, D=1024.  out[b][e][s] = softmax(QK^T/sqrt(D)) V, transposed.
// ---------------------------------------------------------------------------

#define B_BATCH 4
#define S_LEN   2048
#define D_DIM   1024
#define N_ROWS  (B_BATCH * S_LEN)   // 8192
#define CHUNK   128                 // keys per streaming chunk
#define WROW    40                  // LDS row stride (elements) for W tiles (80B)

typedef __attribute__((ext_vector_type(16))) __bf16 v16bf;
typedef __attribute__((ext_vector_type(8)))  __bf16 v8bf;
typedef __attribute__((ext_vector_type(8)))  float  v8f;

// D = A(16x32 bf16) * B(32x16 bf16) + C(16x16 f32)
#define WMMA_BF16(a, b, c) \
  __builtin_amdgcn_wmma_f32_16x16x32_bf16(false, (a), false, (b), (short)0, (c), false, false)

// Load a 16-bit WMMA fragment slice for this lane: two contiguous 8-element
// bf16 runs at p[0..7] (K = hi*8 .. hi*8+7) and p[16..23] (K = hi*8+16 ..).
__device__ inline v16bf load_frag(const __bf16* p) {
  v8bf lo = *(const v8bf*)(p);
  v8bf hi = *(const v8bf*)(p + 16);
  v16bf r;
#pragma unroll
  for (int i = 0; i < 8; ++i) { r[i] = lo[i]; r[i + 8] = hi[i]; }
  return r;
}

// Generic-pointer low 32 bits == LDS byte offset (LDS aperture truncation).
__device__ inline unsigned lds_off(const void* p) {
  return (unsigned)(size_t)p;
}

// Async DMA of one 16B chunk: global -> LDS, tracked by ASYNCcnt.
__device__ inline void async_copy16(unsigned lds, const void* gaddr) {
  asm volatile("global_load_async_to_lds_b128 %0, %1, off"
               :: "v"(lds), "v"(gaddr)
               : "memory");
}
__device__ inline void wait_async() {
  asm volatile("s_wait_asynccnt 0" ::: "memory");
}

// ---------------------------------------------------------------------------
// Kernel 0: fp32 -> bf16 cast (x and the three weight matrices)
// ---------------------------------------------------------------------------
__global__ void cast_bf16_kernel(const float* __restrict__ src,
                                 unsigned short* __restrict__ dst, int n) {
  int i = blockIdx.x * blockDim.x + threadIdx.x;
  if (i < n) {
    __bf16 b = (__bf16)src[i];
    dst[i] = __builtin_bit_cast(unsigned short, b);
  }
}

// ---------------------------------------------------------------------------
// Kernel 1: fused QKV projection.  grid = (N/64, D/64, 3), block = 128.
// mat 0 -> Q[b][s][e], mat 1 -> K[b][s][e], mat 2 -> V^T[b][e][s] (bf16).
// W k-slices (64 rows x 32 k, shared by all 4 waves) are double-buffered in
// LDS via async global->LDS DMA overlapped with the WMMAs.
// ---------------------------------------------------------------------------
__device__ inline void stage_w(unsigned sWbase, int buf, const __bf16* W,
                               int e0, int kt, int tid) {
  // 64 rows x 64 bytes = 256 chunks of 16B; 128 threads -> exactly 2 each.
#pragma unroll
  for (int i = 0; i < 2; ++i) {
    const int c = tid + i * 128;
    const int row = c >> 2, quad = c & 3;
    const unsigned lds = sWbase +
        (unsigned)(buf * (64 * WROW * 2) + row * (WROW * 2) + quad * 16);
    const char* g = (const char*)(W + (size_t)(e0 + row) * D_DIM + kt) + quad * 16;
    async_copy16(lds, g);
  }
}

__global__ void __launch_bounds__(128)
qkv_kernel(const unsigned short* __restrict__ Xus,
           const unsigned short* __restrict__ Wus,
           unsigned short* __restrict__ Qus,
           unsigned short* __restrict__ Kus,
           unsigned short* __restrict__ Vtus) {
  __shared__ __bf16 sW[2][64 * WROW];   // double-buffered W tile, padded rows

  const __bf16* Xbf = (const __bf16*)Xus;
  const int tid  = threadIdx.x;
  const int lane = tid & 31;
  const int wave = tid >> 5;
  const int ln15 = lane & 15;
  const int hi   = lane >> 4;

  const int m0  = blockIdx.x * 64 + wave * 16;   // output row tile (global row)
  const int e0  = blockIdx.y * 64;               // output col tile (4 x 16)
  const int mat = blockIdx.z;                    // 0=Q 1=K 2=V

  const __bf16* W = (const __bf16*)Wus + (size_t)mat * D_DIM * D_DIM;
  const unsigned sWbase = lds_off(&sW[0][0]);

  v8f z = {};
  v8f acc[4];
#pragma unroll
  for (int t = 0; t < 4; ++t) acc[t] = z;

  const __bf16* arow = Xbf + (size_t)(m0 + ln15) * D_DIM + hi * 8;

  // Prologue: stage k-tile 0.
  stage_w(sWbase, 0, W, e0, 0, tid);
  wait_async();
  __syncthreads();

  // Steady state: stage next tile while computing current.  Keep the loop
  // rolled (unroll 1) so the accumulator registers stay tied in place.
  int buf = 0;
#pragma unroll 1
  for (int kt = 0; kt < D_DIM; kt += 32) {
    if (kt + 32 < D_DIM) stage_w(sWbase, buf ^ 1, W, e0, kt + 32, tid);
    __builtin_prefetch(arow + kt + 64, 0, 3);
    v16bf a = load_frag(arow + kt);
    const __bf16* sWbuf = &sW[buf][0];
    // Preload all four B fragments into distinct registers so the ds_loads
    // pipeline behind one wait, then issue the WMMAs back-to-back.
    v16bf bm[4];
#pragma unroll
    for (int t = 0; t < 4; ++t)
      bm[t] = load_frag(sWbuf + (t * 16 + ln15) * WROW + hi * 8);
#pragma unroll
    for (int t = 0; t < 4; ++t)
      acc[t] = WMMA_BF16(a, bm[t], acc[t]);
    wait_async();
    __syncthreads();
    buf ^= 1;
  }

  if (mat == 2) {
    // V stored transposed: Vt[b][e][s]; each lane's 8 C rows are contiguous s.
    const int bidx = m0 >> 11;            // m0 / S_LEN
    const int sb   = m0 & (S_LEN - 1);
#pragma unroll
    for (int t = 0; t < 4; ++t) {
      const int e = e0 + t * 16 + ln15;
      __bf16* vp = (__bf16*)Vtus + ((size_t)bidx * D_DIM + e) * S_LEN + sb + 8 * hi;
      v8bf pk;
#pragma unroll
      for (int r = 0; r < 8; ++r) pk[r] = (__bf16)acc[t][r];
      *(v8bf*)vp = pk;
    }
  } else {
    __bf16* O = (__bf16*)(mat == 0 ? Qus : Kus);
#pragma unroll
    for (int t = 0; t < 4; ++t) {
      const int e = e0 + t * 16 + ln15;
#pragma unroll
      for (int r = 0; r < 8; ++r)
        O[(size_t)(m0 + r + 8 * hi) * D_DIM + e] = (__bf16)acc[t][r];
    }
  }
}

// ---------------------------------------------------------------------------
// Kernel 2: streaming-softmax attention.  grid = (S/16, B), block = 256.
// The 16x1024 Q tile (contiguous 32KB) is async-DMA'd into LDS once and the
// score-GEMM A-fragments are served from LDS for all 8 waves / 16 chunks.
// ---------------------------------------------------------------------------
__global__ void __launch_bounds__(256)
attn_kernel(const unsigned short* __restrict__ Qus,
            const unsigned short* __restrict__ Kus,
            const unsigned short* __restrict__ Vtus,
            float* __restrict__ out) {
  __shared__ __bf16 sQ[16 * D_DIM];    // 32 KB: the block's query tile
  __shared__ float  sS[16][CHUNK];     // raw scores (fp32)
  __shared__ __bf16 sP[16][CHUNK];     // exp(scores) as bf16 (WMMA A operand)
  __shared__ float  red[16][16];       // per-row reduction partials
  __shared__ float  rowMax[16], rowSum[16], rowScale[16];

  const int tid  = threadIdx.x;
  const int lane = tid & 31;
  const int wave = tid >> 5;
  const int ln15 = lane & 15;
  const int hi   = lane >> 4;
  const int s0   = blockIdx.x * 16;
  const int b    = blockIdx.y;

  const __bf16* Qb = (const __bf16*)Qus + (size_t)b * S_LEN * D_DIM;
  const __bf16* Kb = (const __bf16*)Kus + (size_t)b * S_LEN * D_DIM;
  const __bf16* Vb = (const __bf16*)Vtus + (size_t)b * D_DIM * S_LEN;

  // Async-stage the contiguous 32KB Q tile into LDS (8 x 16B per thread).
  {
    const char* qtile = (const char*)(Qb + (size_t)s0 * D_DIM);
    const unsigned sQb = lds_off(sQ);
#pragma unroll
    for (int i = 0; i < (16 * D_DIM * 2) / (256 * 16); ++i) {
      const int off = tid * 16 + i * (256 * 16);
      async_copy16(sQb + off, qtile + off);
    }
  }

  if (tid < 16) { rowMax[tid] = -3.0e38f; rowSum[tid] = 0.0f; }

  v8f z = {};
  v8f acc[8];
#pragma unroll
  for (int t = 0; t < 8; ++t) acc[t] = z;

  const float scale = 0.03125f;  // 1/sqrt(1024)
  const int row = tid >> 4, sub = tid & 15;

  wait_async();
  const __bf16* qrow = sQ + ln15 * D_DIM + hi * 8;

  for (int c0 = 0; c0 < S_LEN; c0 += CHUNK) {
    __syncthreads();

    // ---- Phase 1: this wave's 16x16 score tile (keys j0..j0+15) ----
    const int j0 = c0 + wave * 16;
    const __bf16* krow = Kb + (size_t)(j0 + ln15) * D_DIM + hi * 8;
    v8f cs = z;
#pragma unroll 2
    for (int kt = 0; kt < D_DIM; kt += 32) {
      v16bf a  = load_frag(qrow + kt);        // ds_load from staged Q
      v16bf bm = load_frag(krow + kt);        // global_load (per-wave keys)
      cs = WMMA_BF16(a, bm, cs);
    }
#pragma unroll
    for (int r = 0; r < 8; ++r)
      sS[r + 8 * hi][wave * 16 + ln15] = cs[r] * scale;
    __syncthreads();

    // ---- Phase 2: online softmax stats ----
    float pm = -3.0e38f;
#pragma unroll
    for (int q = 0; q < 8; ++q) pm = fmaxf(pm, sS[row][sub * 8 + q]);
    red[row][sub] = pm;
    __syncthreads();
    if (tid < 16) {
      float cm = red[tid][0];
      for (int q = 1; q < 16; ++q) cm = fmaxf(cm, red[tid][q]);
      float nm = fmaxf(rowMax[tid], cm);
      rowScale[tid] = __expf(rowMax[tid] - nm);
      rowSum[tid]  *= rowScale[tid];
      rowMax[tid]   = nm;
    }
    __syncthreads();
    {
      float nm = rowMax[row];
      float ps = 0.0f;
#pragma unroll
      for (int q = 0; q < 8; ++q) {
        float e = __expf(sS[row][sub * 8 + q] - nm);
        sP[row][sub * 8 + q] = (__bf16)e;
        ps += e;
      }
      red[row][sub] = ps;
    }
    __syncthreads();
    if (tid < 16) {
      float s = 0.0f;
      for (int q = 0; q < 16; ++q) s += red[tid][q];
      rowSum[tid] += s;
    }

    // ---- Phase 3: rescale running accumulators, then P @ V ----
#pragma unroll
    for (int r = 0; r < 8; ++r) {
      float f = rowScale[r + 8 * hi];
#pragma unroll
      for (int t = 0; t < 8; ++t) acc[t][r] *= f;
    }
#pragma unroll
    for (int kt = 0; kt < CHUNK; kt += 32) {
      v16bf a = load_frag(&sP[ln15][kt + hi * 8]);
      // Two groups of 4: issue 4 global B-loads, one wait, 4 WMMAs.
#pragma unroll
      for (int tg = 0; tg < 8; tg += 4) {
        v16bf bm[4];
#pragma unroll
        for (int u = 0; u < 4; ++u) {
          const int e = wave * 128 + (tg + u) * 16 + ln15;
          bm[u] = load_frag(Vb + (size_t)e * S_LEN + c0 + kt + hi * 8);
        }
#pragma unroll
        for (int u = 0; u < 4; ++u)
          acc[tg + u] = WMMA_BF16(a, bm[u], acc[tg + u]);
      }
    }
    __syncthreads();
  }

  // ---- Epilogue: normalize and store transposed output [b][e][s] ----
  float invr[8];
#pragma unroll
  for (int r = 0; r < 8; ++r) invr[r] = 1.0f / rowSum[r + 8 * hi];
#pragma unroll
  for (int t = 0; t < 8; ++t) {
    const int e = wave * 128 + t * 16 + ln15;
    float* op = out + ((size_t)b * D_DIM + e) * S_LEN + s0 + 8 * hi;
#pragma unroll
    for (int r = 0; r < 8; ++r) op[r] = acc[t][r] * invr[r];
  }
}

// ---------------------------------------------------------------------------
// Launch
// ---------------------------------------------------------------------------
extern "C" void kernel_launch(void* const* d_in, const int* in_sizes, int n_in,
                              void* d_out, int out_size, void* d_ws, size_t ws_size,
                              hipStream_t stream) {
  const float* x  = (const float*)d_in[0];
  const float* Wq = (const float*)d_in[1];
  const float* Wk = (const float*)d_in[2];
  const float* Wv = (const float*)d_in[3];
  float* out = (float*)d_out;

  // Workspace layout (bf16 stored as unsigned short):
  //   Xbf : N*D   Wbf : 3*D*D   Qbf : N*D   Kbf : N*D   Vtbf : N*D ([b][e][s])
  unsigned short* Xbf  = (unsigned short*)d_ws;
  unsigned short* Wbf  = Xbf + (size_t)N_ROWS * D_DIM;
  unsigned short* Qbf  = Wbf + (size_t)3 * D_DIM * D_DIM;
  unsigned short* Kbf  = Qbf + (size_t)N_ROWS * D_DIM;
  unsigned short* Vtbf = Kbf + (size_t)N_ROWS * D_DIM;

  const int TPB = 256;
  const int nx = N_ROWS * D_DIM;      // 8388608
  const int nw = D_DIM * D_DIM;       // 1048576
  cast_bf16_kernel<<<(nx + TPB - 1) / TPB, TPB, 0, stream>>>(x,  Xbf, nx);
  cast_bf16_kernel<<<(nw + TPB - 1) / TPB, TPB, 0, stream>>>(Wq, Wbf + 0ull * nw, nw);
  cast_bf16_kernel<<<(nw + TPB - 1) / TPB, TPB, 0, stream>>>(Wk, Wbf + 1ull * nw, nw);
  cast_bf16_kernel<<<(nw + TPB - 1) / TPB, TPB, 0, stream>>>(Wv, Wbf + 2ull * nw, nw);

  dim3 gQKV(N_ROWS / 64, D_DIM / 64, 3);
  qkv_kernel<<<gQKV, 128, 0, stream>>>(Xbf, Wbf, Qbf, Kbf, Vtbf);

  dim3 gATT(S_LEN / 16, B_BATCH);
  attn_kernel<<<gATT, 256, 0, stream>>>(Qbf, Kbf, Vtbf, out);
}
